// MultiHeadAttention_11098195493207
// MI455X (gfx1250) — matrix operand order repaired and checked by reading, compiled, not run on hardware
//
#include <hip/hip_runtime.h>
#include <math.h>

// ---------------------------------------------------------------------------
// MI455X (gfx1250) multi-head attention: bf16 WMMA projections + flash attn
// with async-to-LDS double-buffered K/V tiles.
// ---------------------------------------------------------------------------

#define B_ 2
#define S_ 1024
#define C_ 1024
#define D_ 1024
#define H_ 16
#define E_ 64

typedef __bf16 bf16_t;
typedef __attribute__((ext_vector_type(16))) __bf16 v16bf;
typedef __attribute__((ext_vector_type(8)))  float  v8f;

union FragB16 { v16bf v; unsigned int u[8]; };

// A-matrix (16x32, 16-bit) K-pair base for dword p, per ISA 7.12.2:
// lanes 0-15: VGPR0..3 -> K=0..7, VGPR4..7 -> K=16..23 ; lanes 16-31: +8
__device__ __forceinline__ int kA0(int p, int hi) {
    return (p < 4 ? 2 * p : 16 + 2 * (p - 4)) + (hi ? 8 : 0);
}
// B-matrix (32x16, 16-bit): same K order as A (VGPR0..3 -> K=0..7, VGPR4..7 -> K=16..23; lanes 16-31: +8)
__device__ __forceinline__ int kB0(int p, int hi) {
    return (p < 4 ? 2 * p : 16 + 2 * (p - 4)) + (hi ? 8 : 0);
}

// ---------------------------------------------------------------------------
// Kernel 1: per-head projections  X[1024,1024] @ W[1024,64] (+bias, act)
//   which==0: Q ;  which==1: K*(1/sqrt(E)) ;  which==2: tanh(V)
// Output bf16 row-major [bh][row][e] into workspace.
// ---------------------------------------------------------------------------
__global__ __launch_bounds__(128)
void proj_kernel(const float* __restrict__ seq, const float* __restrict__ ctx,
                 const float* __restrict__ Wq, const float* __restrict__ Wk,
                 const float* __restrict__ Wv, const float* __restrict__ bq,
                 const float* __restrict__ bk, const float* __restrict__ bv,
                 bf16_t* __restrict__ qb, bf16_t* __restrict__ kb,
                 bf16_t* __restrict__ vb) {
    const int mt = blockIdx.x, bh = blockIdx.y, which = blockIdx.z;
    const int b = bh >> 4, h = bh & 15;
    const float* X = (which == 0) ? (seq + (size_t)b * S_ * D_)
                                  : (ctx + (size_t)b * C_ * D_);
    const float* W = ((which == 0) ? Wq : (which == 1) ? Wk : Wv) + (size_t)h * D_ * E_;
    const float* bias = ((which == 0) ? bq : (which == 1) ? bk : bv) + h * E_;
    bf16_t* out = ((which == 0) ? qb : (which == 1) ? kb : vb) + (size_t)bh * S_ * E_;

    const int tid = threadIdx.x;
    const int wave = tid >> 5, lane = tid & 31, laneLo = lane & 15, hi = lane >> 4;

    __shared__ bf16_t As[64][36];   // [m][k]   row-major, pair-aligned
    __shared__ bf16_t Bs[64][36];   // [e][k]   W transposed, pair-aligned

    v8f acc[4] = {};

    for (int kk = 0; kk < D_; kk += 32) {
        __syncthreads();
        // 64x32 fp32 -> bf16 A tile (coalesced reads)
        #pragma unroll
        for (int i = tid; i < 2048; i += 128) {
            int m = i >> 5, k = i & 31;
            As[m][k] = (bf16_t)X[(size_t)(mt * 64 + m) * D_ + kk + k];
        }
        // 32x64 W chunk, stored transposed [e][k] (coalesced reads)
        #pragma unroll
        for (int i = tid; i < 2048; i += 128) {
            int kl = i >> 6, e = i & 63;
            Bs[e][kl] = (bf16_t)W[(size_t)(kk + kl) * E_ + e];
        }
        __syncthreads();

        FragB16 a;
        const int m = wave * 16 + laneLo;
        #pragma unroll
        for (int p = 0; p < 8; ++p)
            a.u[p] = *(const unsigned int*)&As[m][kA0(p, hi)];

        #pragma unroll
        for (int t = 0; t < 4; ++t) {
            FragB16 bb;
            const int n = t * 16 + laneLo;
            #pragma unroll
            for (int p = 0; p < 8; ++p)
                bb.u[p] = *(const unsigned int*)&Bs[n][kB0(p, hi)];
            acc[t] = __builtin_amdgcn_wmma_f32_16x16x32_bf16(
                false, a.v, false, bb.v, (short)0, acc[t], false, false);
        }
    }

    // epilogue: bias + per-projection activation/scale, store bf16
    #pragma unroll
    for (int t = 0; t < 4; ++t) {
        const int n = t * 16 + laneLo;
        const float bsv = bias[n];
        #pragma unroll
        for (int r = 0; r < 8; ++r) {
            const int row = mt * 64 + wave * 16 + r + hi * 8;
            float v = acc[t][r] + bsv;
            if (which == 1) v *= 0.125f;           // 1/sqrt(E)
            else if (which == 2) v = tanhf(v);     // tanh(V)
            out[(size_t)row * E_ + n] = (bf16_t)v;
        }
    }
}

// ---------------------------------------------------------------------------
// Kernel 2: flash attention per (b,h). 64-row Q block per workgroup (4 waves,
// 16 rows each). Double-buffered causal K/V tiles; K tiles filled with
// GLOBAL_LOAD_ASYNC_TO_LDS_B128 (ASYNCcnt), V tiles manually transposed.
// ---------------------------------------------------------------------------
__global__ __launch_bounds__(128)
void attn_kernel(const bf16_t* __restrict__ qb, const bf16_t* __restrict__ kb,
                 const bf16_t* __restrict__ vb, float* __restrict__ out) {
    const int it = blockIdx.x, bh = blockIdx.y;
    const int b = bh >> 4, h = bh & 15;
    const bf16_t* Q = qb + (size_t)bh * S_ * E_;
    const bf16_t* K = kb + (size_t)bh * C_ * E_;
    const bf16_t* V = vb + (size_t)bh * C_ * E_;

    const int tid = threadIdx.x;
    const int wave = tid >> 5, lane = tid & 31, laneLo = lane & 15, hi = lane >> 4;

    // Ks row stride = 72 elems = 144 B (multiple of 16 B for async b128 chunks)
    __shared__ bf16_t Ks[2][64][72];    // [c][e]  (async copy, layout-preserving)
    __shared__ bf16_t Vs[2][64][68];    // [e][c]  (manual transposed copy)
    __shared__ bf16_t Ps[4][16][68];    // per-wave P tile for A re-pack

    // ---- async K-tile fill: 8 KB as 512 x b128 lane-chunks (4 per thread)
    auto asyncCopyK = [&](int jt, int buf) {
        const unsigned ldsbase = (unsigned)(size_t)&Ks[buf][0][0];
        const unsigned long long gbase =
            (unsigned long long)(const void*)(K + (size_t)jt * 64 * E_);
        #pragma unroll
        for (int i = 0; i < 4; ++i) {
            const int idx = tid + i * 128;          // 0..511
            const int row = idx >> 3, ch = idx & 7; // 64 rows x 8 chunks of 16B
            const unsigned lds  = ldsbase + (unsigned)(row * 144 + ch * 16);
            const unsigned voff = (unsigned)(row * 128 + ch * 16);
            asm volatile("global_load_async_to_lds_b128 %0, %1, %2"
                         :: "v"(lds), "v"(voff), "s"(gbase) : "memory");
        }
    };
    auto copyV = [&](int jt, int buf) {
        for (int i = tid; i < 4096; i += 128) {
            int c = i >> 6, e = i & 63;
            Vs[buf][e][c] = V[(size_t)(jt * 64 + c) * E_ + e];
        }
    };

    // Q strip fragments (16 rows x 64 e) kept in registers for whole block
    FragB16 qa[2];
    {
        const int qrow = it * 64 + wave * 16 + laneLo;
        #pragma unroll
        for (int ch = 0; ch < 2; ++ch)
            #pragma unroll
            for (int p = 0; p < 8; ++p)
                qa[ch].u[p] = *(const unsigned int*)&Q[(size_t)qrow * E_ + ch * 32 + kA0(p, hi)];
    }

    float mrow[8], lrow[8];
    v8f o[4] = {};
    #pragma unroll
    for (int r = 0; r < 8; ++r) { mrow[r] = -INFINITY; lrow[r] = 0.f; }

    // ---- preload tile 0
    asyncCopyK(0, 0);
    copyV(0, 0);
    asm volatile("s_wait_asynccnt 0" ::: "memory");
    __syncthreads();

    for (int jt = 0; jt <= it; ++jt) {      // causal: only tiles with jt<=it
        const int cur = jt & 1;
        // prefetch next tile into the other buffer (overlaps with compute)
        if (jt < it) {
            asyncCopyK(jt + 1, cur ^ 1);
            copyV(jt + 1, cur ^ 1);
        }

        // S = Q * K^T  (scale already folded into K)
        v8f s[4] = {};
        #pragma unroll
        for (int t = 0; t < 4; ++t) {
            const int n = t * 16 + laneLo;        // local column c
            #pragma unroll
            for (int ch = 0; ch < 2; ++ch) {
                FragB16 kf;
                #pragma unroll
                for (int p = 0; p < 8; ++p)
                    kf.u[p] = *(const unsigned int*)&Ks[cur][n][ch * 32 + kB0(p, hi)];
                s[t] = __builtin_amdgcn_wmma_f32_16x16x32_bf16(
                    false, qa[ch].v, false, kf.v, (short)0, s[t], false, false);
            }
        }

        // causal mask on diagonal tile
        if (jt == it) {
            #pragma unroll
            for (int t = 0; t < 4; ++t) {
                const int gcol = jt * 64 + t * 16 + laneLo;
                #pragma unroll
                for (int r = 0; r < 8; ++r) {
                    const int grow = it * 64 + wave * 16 + r + hi * 8;
                    if (gcol > grow) s[t][r] = -INFINITY;
                }
            }
        }

        // online softmax (row groups live across 16 lanes -> xor reductions)
        float newm[8], fac[8];
        #pragma unroll
        for (int r = 0; r < 8; ++r) {
            float v = fmaxf(fmaxf(s[0][r], s[1][r]), fmaxf(s[2][r], s[3][r]));
            #pragma unroll
            for (int off = 1; off < 16; off <<= 1)
                v = fmaxf(v, __shfl_xor(v, off, 32));
            newm[r] = fmaxf(mrow[r], v);
            fac[r]  = __expf(mrow[r] - newm[r]);
            mrow[r] = newm[r];
        }

        v8f p4[4];
        #pragma unroll
        for (int t = 0; t < 4; ++t)
            #pragma unroll
            for (int r = 0; r < 8; ++r)
                p4[t][r] = __expf(s[t][r] - newm[r]);

        #pragma unroll
        for (int r = 0; r < 8; ++r) {
            float v = p4[0][r] + p4[1][r] + p4[2][r] + p4[3][r];
            #pragma unroll
            for (int off = 1; off < 16; off <<= 1)
                v += __shfl_xor(v, off, 32);
            lrow[r] = lrow[r] * fac[r] + v;
        }

        #pragma unroll
        for (int t = 0; t < 4; ++t)
            #pragma unroll
            for (int r = 0; r < 8; ++r)
                o[t][r] *= fac[r];

        // spill P (bf16) so it can be re-gathered in A-fragment layout
        #pragma unroll
        for (int t = 0; t < 4; ++t)
            #pragma unroll
            for (int r = 0; r < 8; ++r)
                Ps[wave][r + hi * 8][t * 16 + laneLo] = (bf16_t)p4[t][r];
        __syncthreads();

        FragB16 pa[2];
        #pragma unroll
        for (int ch = 0; ch < 2; ++ch)
            #pragma unroll
            for (int p = 0; p < 8; ++p)
                pa[ch].u[p] = *(const unsigned int*)&Ps[wave][laneLo][ch * 32 + kA0(p, hi)];

        // O += P * V
        #pragma unroll
        for (int t = 0; t < 4; ++t) {
            const int e = t * 16 + laneLo;
            #pragma unroll
            for (int ch = 0; ch < 2; ++ch) {
                FragB16 vf;
                #pragma unroll
                for (int p = 0; p < 8; ++p)
                    vf.u[p] = *(const unsigned int*)&Vs[cur][e][ch * 32 + kB0(p, hi)];
                o[t] = __builtin_amdgcn_wmma_f32_16x16x32_bf16(
                    false, pa[ch].v, false, vf.v, (short)0, o[t], false, false);
            }
        }

        // drain prefetch, then make all waves' fills visible before next iter
        if (jt < it)
            asm volatile("s_wait_asynccnt 0" ::: "memory");
        __syncthreads();
    }

    // normalize and store fp32 output [B, S, H*E]
    #pragma unroll
    for (int t = 0; t < 4; ++t) {
        const int e = t * 16 + laneLo;
        #pragma unroll
        for (int r = 0; r < 8; ++r) {
            const int srow = it * 64 + wave * 16 + r + hi * 8;
            out[((size_t)b * S_ + srow) * (H_ * E_) + h * E_ + e] = o[t][r] / lrow[r];
        }
    }
}

// ---------------------------------------------------------------------------
extern "C" void kernel_launch(void* const* d_in, const int* in_sizes, int n_in,
                              void* d_out, int out_size, void* d_ws, size_t ws_size,
                              hipStream_t stream) {
    (void)in_sizes; (void)n_in; (void)out_size; (void)ws_size;
    const float* seq = (const float*)d_in[0];
    const float* ctx = (const float*)d_in[1];
    const float* Wq  = (const float*)d_in[2];
    const float* Wk  = (const float*)d_in[3];
    const float* Wv  = (const float*)d_in[4];
    const float* bq  = (const float*)d_in[5];
    const float* bk  = (const float*)d_in[6];
    const float* bv  = (const float*)d_in[7];
    float* out = (float*)d_out;

    const size_t per = (size_t)B_ * H_ * S_ * E_;   // 2,097,152 elems (4 MB bf16)
    bf16_t* qb = (bf16_t*)d_ws;
    bf16_t* kb = qb + per;
    bf16_t* vb = kb + per;

    dim3 blk(128);
    dim3 g1(S_ / 64, B_ * H_, 3);
    proj_kernel<<<g1, blk, 0, stream>>>(seq, ctx, Wq, Wk, Wv, bq, bk, bv, qb, kb, vb);

    dim3 g2(S_ / 64, B_ * H_);
    attn_kernel<<<g2, blk, 0, stream>>>(qb, kb, vb, out);
}
